// PyramidSparseAttention_31258771980507
// MI455X (gfx1250) — matrix operand order repaired and checked by reading, not compiled, not run on hardware
//
#include <hip/hip_runtime.h>
#include <hip/hip_bf16.h>

typedef __attribute__((ext_vector_type(16))) _Float16 v16h;
typedef __attribute__((ext_vector_type(8)))  _Float16 v8h;
typedef __attribute__((ext_vector_type(8)))  float    v8f;
typedef __attribute__((__vector_size__(16))) int      i32x4;

#define HN 8
#define LN 4096
#define DN 64
#define NB 64   /* number of 64-wide blocks along L */

// D = A*B + C, 16x16x32 f16 -> f32
__device__ __forceinline__ v8f wmma_f16(v16h a, v16h b, v8f c) {
  return __builtin_amdgcn_wmma_f32_16x16x32_f16(false, a, false, b, (short)0, c,
                                                false, false);
}

// Async global -> LDS copy of 16 bytes (GLOBAL_LOAD_ASYNC_TO_LDS_B128, ASYNCcnt).
__device__ __forceinline__ void async_b128(const _Float16* gsrc, _Float16* ldst) {
#if __has_builtin(__builtin_amdgcn_global_load_async_to_lds_b128)
  __builtin_amdgcn_global_load_async_to_lds_b128(
      (i32x4*)(void*)gsrc, (i32x4*)(void*)ldst, 0, 0);
#else
  asm volatile("global_load_async_to_lds_b128 %0, %1, off"
               :: "v"((unsigned)(uintptr_t)ldst), "v"(gsrc)
               : "memory");
#endif
}

__device__ __forceinline__ void wait_async0() {
#if __has_builtin(__builtin_amdgcn_s_wait_asynccnt)
  __builtin_amdgcn_s_wait_asynccnt(0);
#else
  asm volatile("s_wait_asynccnt 0x0" ::: "memory");
#endif
}

// A-matrix 16x32 f16 loader. Lane L: row m = L&15; chunk base = 16*(L>=16).
// Element e -> K = base+e.
// ld = row stride in halves (multiple of 8 so both 8-half chunks stay 16B aligned).
__device__ __forceinline__ v16h loadA16(const _Float16* base, int ld) {
  int lane = threadIdx.x & 31;
  const _Float16* p = base + (lane & 15) * ld + ((lane >> 4) << 4);
  v8h lo = *(const v8h*)(p);
  v8h hi = *(const v8h*)(p + 8);
  return __builtin_shufflevector(lo, hi, 0,1,2,3,4,5,6,7,8,9,10,11,12,13,14,15);
}

// 16 contiguous halves (two 16B reads) -> v16h. p must be 16B aligned.
__device__ __forceinline__ v16h ld16h(const _Float16* p) {
  v8h lo = *(const v8h*)(p);
  v8h hi = *(const v8h*)(p + 8);
  return __builtin_shufflevector(lo, hi, 0,1,2,3,4,5,6,7,8,9,10,11,12,13,14,15);
}

// ---------------------------------------------------------------------------
// Kernel 0: convert K to f16 row-major and V to f16 block-transposed layout so
// the attention kernel can fill LDS with async b128 copies (no VGPR roundtrip).
//   kh [h][t][d]          (f16)
//   vth[h][kb][d][n] = v[h][kb*64+n][d]   (f16)
// grid = 8*4096 blocks of 64 threads; one block = one token row.
// ---------------------------------------------------------------------------
__global__ void psa_conv(const float* __restrict__ k, const float* __restrict__ v,
                         _Float16* __restrict__ kh, _Float16* __restrict__ vth) {
  int bx = blockIdx.x;
  int h = bx >> 12;
  int t = bx & 4095;
  int kb = t >> 6, n = t & 63;
  int d = threadIdx.x;
  int g = ((h * LN) + t) * DN + d;
  kh[g] = (_Float16)k[g];
  vth[(((h * NB) + kb) * DN + d) * 64 + n] = (_Float16)v[g];
}

// ---------------------------------------------------------------------------
// Kernel 1: gather sampled rows of q (pre-scaled by 1/sqrt(D)) and k into f16.
// grid = 2*8*1024 blocks of 64 threads; one block = one sampled row.
// ---------------------------------------------------------------------------
__global__ void psa_gather(const float* __restrict__ q, const float* __restrict__ k,
                           const int* __restrict__ siq, const int* __restrict__ sik,
                           _Float16* __restrict__ sqh, _Float16* __restrict__ skh) {
  int task  = blockIdx.x;
  int which = task >> 13;       // 0 = q, 1 = k
  int rem   = task & 8191;
  int h     = rem >> 10;
  int s     = rem & 1023;       // sampled-row index within head (64 blocks * 16)
  int blk   = s >> 4;
  int j     = s & 15;
  int d     = threadIdx.x;
  if (which == 0) {
    int p = siq[h * 16 + j];
    sqh[((h << 10) + s) * DN + d] =
        (_Float16)(q[((h * LN) + blk * 64 + p) * DN + d] * 0.125f);
  } else {
    int p = sik[h * 16 + j];
    skh[((h << 10) + s) * DN + d] =
        (_Float16)(k[((h * LN) + blk * 64 + p) * DN + d]);
  }
}

// ---------------------------------------------------------------------------
// Kernel 2: block mask. grid = (64 qb, 8 h), 128 threads (4 waves).
// ---------------------------------------------------------------------------
__global__ __launch_bounds__(128) void psa_mask(const _Float16* __restrict__ sqh,
                                                const _Float16* __restrict__ skh,
                                                int* __restrict__ maskbuf) {
  __shared__ _Float16 sc[16][1024];     // 32 KB score tile (f16)
  __shared__ float rowmax[16], rowsum[16], pooled[64];

  int qb = blockIdx.x, h = blockIdx.y;
  int tid = threadIdx.x, lane = tid & 31, wave = tid >> 5;

  const _Float16* sqp = sqh + ((h << 10) + qb * 16) * DN;
  v16h a0 = loadA16(sqp,      DN);
  v16h a1 = loadA16(sqp + 32, DN);

  int nlo = lane & 15;
  int khalf = (lane >> 4) << 4;         // 0 or 16
  int mb = (lane >> 4) << 3;            // row offset 0 or 8

  for (int kb = wave; kb < NB; kb += 4) {
    const _Float16* skp = skh + ((h << 10) + kb * 16) * DN + nlo * DN;
    v16h b0 = ld16h(skp + khalf);
    v16h b1 = ld16h(skp + 32 + khalf);
    v8f c = {0.f, 0.f, 0.f, 0.f, 0.f, 0.f, 0.f, 0.f};
    c = wmma_f16(a0, b0, c);
    c = wmma_f16(a1, b1, c);
    for (int r = 0; r < 8; ++r)
      sc[mb + r][kb * 16 + nlo] = (_Float16)c[r];
  }
  __syncthreads();

  // softmax stats: 8 consecutive lanes per row, vectorized 16B LDS reads
  int row = tid >> 3, sub = tid & 7;
  const _Float16* rp = &sc[row][0];
  float mx = -3.0e38f;
  for (int c0 = sub * 128; c0 < sub * 128 + 128; c0 += 8) {
    v8h x = *(const v8h*)(rp + c0);
    for (int e = 0; e < 8; ++e) mx = fmaxf(mx, (float)x[e]);
  }
  for (int m2 = 4; m2 >= 1; m2 >>= 1) mx = fmaxf(mx, __shfl_xor(mx, m2, 32));
  float sm = 0.f;
  for (int c0 = sub * 128; c0 < sub * 128 + 128; c0 += 8) {
    v8h x = *(const v8h*)(rp + c0);
    for (int e = 0; e < 8; ++e) sm += __expf((float)x[e] - mx);
  }
  for (int m2 = 4; m2 >= 1; m2 >>= 1) sm += __shfl_xor(sm, m2, 32);
  if (sub == 0) { rowmax[row] = mx; rowsum[row] = sm; }
  __syncthreads();

  // 16x16 sum-pooling of the softmaxed attention per key block
  if (tid < 64) {
    float acc = 0.f;
    for (int r = 0; r < 16; ++r) {
      float rm = rowmax[r], inv = 1.0f / rowsum[r];
      v8h x0 = *(const v8h*)&sc[r][tid * 16];
      v8h x1 = *(const v8h*)&sc[r][tid * 16 + 8];
      for (int e = 0; e < 8; ++e)
        acc += (__expf((float)x0[e] - rm) + __expf((float)x1[e] - rm)) * inv;
    }
    pooled[tid] = acc;
  }
  __syncthreads();

  // keep block iff cumulative mass of strictly-better-ranked blocks < 50%
  if (tid < 64) {
    float pi = pooled[tid];
    float total = 0.f, before = 0.f;
    for (int j = 0; j < 64; ++j) {
      float pj = pooled[j];
      total += pj;
      if (pj > pi || (pj == pi && j < tid)) before += pj;
    }
    maskbuf[(h * NB + qb) * NB + tid] = (before < 0.5f * total) ? 1 : 0;
  }
}

// ---------------------------------------------------------------------------
// Kernel 3: block-sparse flash attention. grid = (64 qb, 8 h), 128 threads.
// Wave w owns rows [w*16, w*16+16). K/V tiles stream in via async LDS copies.
// ---------------------------------------------------------------------------
__global__ __launch_bounds__(128) void psa_attn(const float* __restrict__ q,
                                                const _Float16* __restrict__ kh,
                                                const _Float16* __restrict__ vth,
                                                const int* __restrict__ maskbuf,
                                                float* __restrict__ out) {
  __shared__ _Float16 Qs[64][72];
  __shared__ _Float16 Ks[64][72];   // K row-major [n][d]
  __shared__ _Float16 Vt[64][72];   // V transposed [d][n]
  __shared__ _Float16 Ps[64][72];   // P strip, re-layout staging
  __shared__ int mrow[64];

  int qb = blockIdx.x, h = blockIdx.y;
  int tid = threadIdx.x, lane = tid & 31, wave = tid >> 5;

  for (int i = tid; i < 4096; i += 128) {
    int r = i >> 6, c = i & 63;
    Qs[r][c] = (_Float16)(q[((h * LN) + qb * 64 + r) * DN + c] * 0.125f);
  }
  for (int i = tid; i < 64; i += 128) mrow[i] = maskbuf[(h * NB + qb) * NB + i];
  __syncthreads();

  v16h aQ0 = loadA16(&Qs[wave * 16][0],  72);
  v16h aQ1 = loadA16(&Qs[wave * 16][32], 72);

  const v8f vz = {0.f, 0.f, 0.f, 0.f, 0.f, 0.f, 0.f, 0.f};
  float mrun[8], lrun[8];
  v8f accO[4];
  for (int r = 0; r < 8; ++r) { mrun[r] = -3.0e38f; lrun[r] = 0.f; }
  for (int t = 0; t < 4; ++t) accO[t] = vz;

  int nlo = lane & 15;
  int khalf = (lane >> 4) << 4;     // 0 or 16
  int mb = (lane >> 4) << 3;        // 0 or 8

  for (int kb = 0; kb < NB; ++kb) {
    if (!mrow[kb]) continue;                 // uniform across the workgroup
    __syncthreads();                         // prior reads of Ks/Vt done
    {
      const _Float16* ksrc = kh  + (((h * LN) + kb * 64) * DN);
      const _Float16* vsrc = vth + (((h * NB) + kb) * DN) * 64;
      // 512 x 16B chunks per matrix; 8 async copies per thread total
      for (int c = tid; c < 512; c += 128) {
        int r = c >> 3, col = (c & 7) * 8;
        async_b128(ksrc + r * 64 + col, &Ks[r][col]);
        async_b128(vsrc + r * 64 + col, &Vt[r][col]);
      }
    }
    wait_async0();
    __syncthreads();

    // S strip: 16 x 64 via 8 WMMAs (Q pre-scaled)
    v8f s[4];
    for (int t = 0; t < 4; ++t) {
      const _Float16* kp = &Ks[t * 16 + nlo][0];
      v16h b0 = ld16h(kp + khalf);
      v16h b1 = ld16h(kp + 32 + khalf);
      v8f c = vz;
      c = wmma_f16(aQ0, b0, c);
      c = wmma_f16(aQ1, b1, c);
      s[t] = c;
    }

    // online softmax: VGPR r holds row (mb + r); reduce over the 16-lane half
    for (int r = 0; r < 8; ++r) {
      float mx = fmaxf(fmaxf(s[0][r], s[1][r]), fmaxf(s[2][r], s[3][r]));
      for (int m2 = 8; m2 >= 1; m2 >>= 1) mx = fmaxf(mx, __shfl_xor(mx, m2, 32));
      float mt = fmaxf(mrun[r], mx);
      float alpha = __expf(mrun[r] - mt);
      float ls = 0.f;
      for (int t = 0; t < 4; ++t) {
        float e = __expf(s[t][r] - mt);
        s[t][r] = e;
        ls += e;
      }
      for (int m2 = 8; m2 >= 1; m2 >>= 1) ls += __shfl_xor(ls, m2, 32);
      lrun[r] = lrun[r] * alpha + ls;
      for (int t = 0; t < 4; ++t) accO[t][r] *= alpha;
      mrun[r] = mt;
      for (int t = 0; t < 4; ++t)
        Ps[wave * 16 + mb + r][t * 16 + nlo] = (_Float16)s[t][r];
    }
    __syncthreads();                         // Ps visible / compiler fence

    // PV: out strip += P(16x64) * V(64x64)
    v16h aP0 = loadA16(&Ps[wave * 16][0],  72);
    v16h aP1 = loadA16(&Ps[wave * 16][32], 72);
    for (int t = 0; t < 4; ++t) {
      const _Float16* vp = &Vt[t * 16 + nlo][0];
      v16h b0 = ld16h(vp + khalf);
      v16h b1 = ld16h(vp + 32 + khalf);
      accO[t] = wmma_f16(aP0, b0, accO[t]);
      accO[t] = wmma_f16(aP1, b1, accO[t]);
    }
  }

  for (int t = 0; t < 4; ++t)
    for (int r = 0; r < 8; ++r) {
      int row = qb * 64 + wave * 16 + mb + r;
      out[((h * LN) + row) * DN + t * 16 + nlo] = accO[t][r] / lrun[r];
    }
}

// ---------------------------------------------------------------------------
extern "C" void kernel_launch(void* const* d_in, const int* in_sizes, int n_in,
                              void* d_out, int out_size, void* d_ws, size_t ws_size,
                              hipStream_t stream) {
  const float* q  = (const float*)d_in[0];
  const float* k  = (const float*)d_in[1];
  const float* v  = (const float*)d_in[2];
  const int* siq  = (const int*)d_in[3];
  const int* sik  = (const int*)d_in[4];
  float* out      = (float*)d_out;

  char* ws = (char*)d_ws;
  int*      maskbuf = (int*)ws;                                  // 128 KB
  _Float16* sqh     = (_Float16*)(ws + 131072);                  // 1 MB
  _Float16* skh     = (_Float16*)(ws + 131072 + 1048576);        // 1 MB
  _Float16* kh      = (_Float16*)(ws + 131072 + 2097152);        // 4 MB
  _Float16* vth     = (_Float16*)(ws + 131072 + 2097152 + 4194304); // 4 MB

  psa_conv<<<dim3(HN * LN), 64, 0, stream>>>(k, v, kh, vth);
  psa_gather<<<dim3(2 * HN * 1024), 64, 0, stream>>>(q, k, siq, sik, sqh, skh);
  psa_mask<<<dim3(NB, HN), 128, 0, stream>>>(sqh, skh, maskbuf);
  psa_attn<<<dim3(NB, HN), 128, 0, stream>>>(q, kh, vth, maskbuf, out);
}
